// StructuralSelfAttention_56667798503532
// MI455X (gfx1250) — matrix operand order repaired and checked by reading, not compiled, not run on hardware
//
#include <hip/hip_runtime.h>
#include <math.h>

// ---------------------------------------------------------------------------
// StructuralSelfAttention transformer layer for MI455X (gfx1250, wave32)
// bf16 WMMA (v_wmma_f32_16x16x32_bf16) for all GEMMs, f32 accumulate.
// Flash-style fused attention: score strip kept in LDS, never hits HBM.
// Each wave computes two N-tiles sharing one A operand (halves A traffic).
// ---------------------------------------------------------------------------

typedef unsigned short bf16_t;
typedef __attribute__((ext_vector_type(16))) __bf16 v16bf;
typedef __attribute__((ext_vector_type(8)))  float  v8f;

union ABPack { uint4 q[2]; v16bf v; };

static constexpr int Bn  = 8;
static constexpr int En  = 1024;
static constexpr int Dn  = 256;
static constexpr int Hn  = 8;
static constexpr int HDn = 32;   // head dim == one WMMA K step
static constexpr int FFn = 1024; // 4*D

// round-to-nearest-even f32 -> bf16
__device__ __forceinline__ bf16_t f2bf(float f) {
  union { float f; unsigned u; } v; v.f = f;
  unsigned u = v.u;
  if ((u & 0x7fffffffu) > 0x7f800000u) return (bf16_t)0x7fc0; // NaN
  unsigned r = u + 0x7fffu + ((u >> 16) & 1u);
  return (bf16_t)(r >> 16);
}

// Per-lane operand fetch, contiguous k order (k = 16*half + s), same for A and B.
// arow = A  + (lane&15)*lda + (lane>>4)*16  (row-major A, 16 x K)
// brow = BT + (lane&15)*ldb + (lane>>4)*16  (row-major B-transposed, 16 x K)
__device__ __forceinline__ v16bf load_a(const bf16_t* arow, int k) {
  ABPack a;
  a.q[0] = *(const uint4*)(arow + k);
  a.q[1] = *(const uint4*)(arow + k + 8);
  return a.v;
}
__device__ __forceinline__ v16bf load_b(const bf16_t* brow, int k) {
  ABPack b;
  b.q[0] = *(const uint4*)(brow + k);
  b.q[1] = *(const uint4*)(brow + k + 8);
  return b.v;
}
__device__ __forceinline__ v8f wmma_acc(v16bf a, v16bf b, v8f c) {
  return __builtin_amdgcn_wmma_f32_16x16x32_bf16(false, a, false, b,
                                                 (short)0, c, false, false);
}

// ---------------------------------------------------------------------------
// conversion kernels
// ---------------------------------------------------------------------------
__global__ void k_cvt(const float* __restrict__ in, bf16_t* __restrict__ out, int n) {
  int i = blockIdx.x * 256 + threadIdx.x;
  if (i < n) out[i] = f2bf(in[i]);
}

// out[c][r] = bf16(in[r][c])   (weight transpose for BT operand)
__global__ void k_tcvt(const float* __restrict__ in, bf16_t* __restrict__ out,
                       int rows, int cols) {
  int i = blockIdx.x * 256 + threadIdx.x;
  if (i >= rows * cols) return;
  int r = i / cols, c = i % cols;
  out[(size_t)c * rows + r] = f2bf(in[i]);
}

// ---------------------------------------------------------------------------
// QKV projection: q/k/v = x @ W + b.  V written transposed per head.
// Wave computes a 16x32 strip (2 N-tiles, shared A).
// grid: 512 mtiles * 24 pairs / 8 waves = 1536 blocks
// ---------------------------------------------------------------------------
__global__ __launch_bounds__(256) void k_qkv(
    const bf16_t* __restrict__ xb,
    const bf16_t* __restrict__ WqT, const bf16_t* __restrict__ WkT,
    const bf16_t* __restrict__ WvT,
    const float* __restrict__ bq, const float* __restrict__ bk,
    const float* __restrict__ bv,
    bf16_t* __restrict__ Qb, bf16_t* __restrict__ Kb, bf16_t* __restrict__ Vt) {
  int wv   = threadIdx.x >> 5;
  int lane = threadIdx.x & 31;
  int r15 = lane & 15, half = lane >> 4;
  int tile  = blockIdx.x * 8 + wv;
  int mtile = tile / 24;
  int pr    = tile % 24;
  int which = pr / 8;            // 0=q 1=k 2=v
  int n0    = (pr % 8) * 32;     // 32-wide strip
  int m0    = mtile * 16;

  const bf16_t* WT   = (which == 0) ? WqT : (which == 1) ? WkT : WvT;
  const float*  bias = (which == 0) ? bq  : (which == 1) ? bk  : bv;

  const bf16_t* arow  = xb + (size_t)(m0 + r15) * Dn + half * 16;
  const bf16_t* brow0 = WT + (size_t)(n0 + r15) * Dn + half * 16;
  const bf16_t* brow1 = brow0 + (size_t)16 * Dn;

  v8f acc0 = {}, acc1 = {};
  for (int k = 0; k < Dn; k += 32) {
    v16bf a = load_a(arow, k);
    acc0 = wmma_acc(a, load_b(brow0, k), acc0);
    acc1 = wmma_acc(a, load_b(brow1, k), acc1);
  }

  int d0 = n0 + r15, d1 = d0 + 16;
  float bv0 = bias[d0], bv1 = bias[d1];
  if (which < 2) {
    bf16_t* out = (which == 0) ? Qb : Kb;
#pragma unroll
    for (int r = 0; r < 8; ++r) {
      size_t row = (size_t)(m0 + half * 8 + r) * Dn;
      out[row + d0] = f2bf(acc0[r] + bv0);
      out[row + d1] = f2bf(acc1[r] + bv1);
    }
  } else {
    // transposed store: Vt[b][h][hd][e], 8 contiguous e per lane -> B128
    int b  = m0 >> 10;
    int e0 = (m0 & 1023) + half * 8;
    int h  = n0 >> 5;              // strip == one head
    unsigned p0[4], p1[4];
#pragma unroll
    for (int i = 0; i < 4; ++i) {
      p0[i] = (unsigned)f2bf(acc0[2 * i] + bv0) | ((unsigned)f2bf(acc0[2 * i + 1] + bv0) << 16);
      p1[i] = (unsigned)f2bf(acc1[2 * i] + bv1) | ((unsigned)f2bf(acc1[2 * i + 1] + bv1) << 16);
    }
    uint4 u0; u0.x = p0[0]; u0.y = p0[1]; u0.z = p0[2]; u0.w = p0[3];
    uint4 u1; u1.x = p1[0]; u1.y = p1[1]; u1.z = p1[2]; u1.w = p1[3];
    size_t hb = (size_t)(b * Hn + h) * HDn;
    *(uint4*)(Vt + (hb + r15)      * En + e0) = u0;
    *(uint4*)(Vt + (hb + r15 + 16) * En + e0) = u1;
  }
}

// ---------------------------------------------------------------------------
// Fused attention: one block per (b, h, q-tile of 16).
// S strip [16x1024] lives in LDS; softmax in place; P (bf16) @ V via WMMA.
// grid: B*H*(E/16) = 4096 blocks, 256 threads, ~99KB dynamic LDS
// ---------------------------------------------------------------------------
__global__ __launch_bounds__(256) void k_attn(
    const bf16_t* __restrict__ Qb, const bf16_t* __restrict__ Kb,
    const bf16_t* __restrict__ Vt,
    const int* __restrict__ srel, const unsigned char* __restrict__ kpm,
    const float* __restrict__ sbw_g, bf16_t* __restrict__ attnB) {
  extern __shared__ char smem_raw[];
  float*  S      = (float*)smem_raw;                    // 16*1024 f32 = 64KB
  bf16_t* P      = (bf16_t*)(smem_raw + 65536);         // 16*1024 bf16 = 32KB
  float*  red    = (float*)(smem_raw + 65536 + 32768);  // 256
  float*  rowmax = red + 256;                           // 16
  float*  rowinv = rowmax + 16;                         // 16
  float*  Oacc   = rowinv + 16;                         // 16*32
  float*  sbw    = Oacc + 512;                          // NST*H = 48

  int tid = threadIdx.x, lane = tid & 31, wv = tid >> 5;
  int bid = blockIdx.x;
  int qt = bid & 63, h = (bid >> 6) & 7, b = bid >> 9;
  int q0 = qt * 16;
  int r15 = lane & 15, half = lane >> 4;

  if (tid < 48) sbw[tid] = sbw_g[tid];
  Oacc[tid] = 0.f; Oacc[tid + 256] = 0.f;
  __syncthreads();

  // ---- scores: S = scale * Q Kt + struct bias, masked ----
  // A (Q tile) is invariant across key tiles: load once, 8 WMMAs.
  const float scale = 0.17677669529663687f; // 1/sqrt(32)
  const bf16_t* arow = Qb + ((size_t)(b * En + q0 + r15)) * Dn + h * HDn + half * 16;
  v16bf aQ = load_a(arow, 0);
  for (int kt = wv * 8; kt < wv * 8 + 8; ++kt) {
    const bf16_t* brow =
        Kb + ((size_t)(b * En + kt * 16 + r15)) * Dn + h * HDn + half * 16;
    v8f acc = {};
    acc = wmma_acc(aQ, load_b(brow, 0), acc);
    int kk = kt * 16 + r15;
    bool masked = kpm[b * En + kk] != 0;
#pragma unroll
    for (int r = 0; r < 8; ++r) {
      int m = half * 8 + r;
      int q = q0 + m;
      int rel = srel[((size_t)b * En + q) * En + kk];
      float v = acc[r] * scale + sbw[rel * Hn + h];
      if (masked) v = -INFINITY;
      S[m * 1024 + kk] = v;
    }
  }
  __syncthreads();

  // ---- softmax over 1024 keys: 16 threads per row ----
  int row = tid >> 4, j = tid & 15;
  float mx = -INFINITY;
  for (int i = 0; i < 64; ++i) mx = fmaxf(mx, S[row * 1024 + i * 16 + j]);
  red[tid] = mx;
  __syncthreads();
  if (j == 0) {
    float m2 = red[row * 16];
    for (int t = 1; t < 16; ++t) m2 = fmaxf(m2, red[row * 16 + t]);
    rowmax[row] = m2;
  }
  __syncthreads();
  float rmax = rowmax[row];
  float sum = 0.f;
  for (int i = 0; i < 64; ++i) {
    int c = row * 1024 + i * 16 + j;
    float v = S[c];
    float p = (rmax == -INFINITY) ? 0.f : __expf(v - rmax); // all-masked row -> 0
    S[c] = p; sum += p;
  }
  red[tid] = sum;
  __syncthreads();
  if (j == 0) {
    float s2 = 0.f;
    for (int t = 0; t < 16; ++t) s2 += red[row * 16 + t];
    rowinv[row] = (s2 > 0.f) ? (1.f / s2) : 0.f; // nan_to_num semantics
  }
  __syncthreads();
  float inv = rowinv[row];
  for (int i = 0; i < 64; ++i) {
    int c = row * 1024 + i * 16 + j;
    P[c] = f2bf(S[c] * inv);
  }
  __syncthreads();

  // ---- O[16x32] = P[16x1024] @ V[1024x32] ----
  // Each wave: 4 K-chunks of its k-range, BOTH hd halves share one LDS A pack.
  {
    const bf16_t* arowP = (const bf16_t*)P + r15 * 1024 + half * 16;
    size_t hb = (size_t)(b * Hn + h) * HDn;
    const bf16_t* brow0 = Vt + (hb + r15)      * En + half * 16;
    const bf16_t* brow1 = Vt + (hb + r15 + 16) * En + half * 16;
    v8f acc0 = {}, acc1 = {};
    int kbase = wv * 128;
    for (int c = 0; c < 128; c += 32) {
      int k = kbase + c;
      v16bf a = load_a(arowP, k);
      acc0 = wmma_acc(a, load_b(brow0, k), acc0);
      acc1 = wmma_acc(a, load_b(brow1, k), acc1);
    }
#pragma unroll
    for (int r = 0; r < 8; ++r) {
      int m = half * 8 + r;
      atomicAdd(&Oacc[m * 32 + r15],      acc0[r]);   // ds_add_f32
      atomicAdd(&Oacc[m * 32 + r15 + 16], acc1[r]);
    }
  }
  __syncthreads();
  {
    int rr = tid >> 4, c = tid & 15;
    size_t base = ((size_t)(b * En + q0 + rr)) * Dn + h * HDn;
    attnB[base + c]      = f2bf(Oacc[rr * 32 + c]);
    attnB[base + c + 16] = f2bf(Oacc[rr * 32 + c + 16]);
  }
}

// ---------------------------------------------------------------------------
// Output projection + residual + LN1.  One block per 16 tokens.
// Wave: 16x32 strip (2 N-tiles, shared A).
// ---------------------------------------------------------------------------
__global__ __launch_bounds__(256) void k_proj_ln1(
    const bf16_t* __restrict__ attnB, const bf16_t* __restrict__ WoT,
    const float* __restrict__ bo, const float* __restrict__ x,
    const float* __restrict__ g1, const float* __restrict__ b1,
    float* __restrict__ x1f, bf16_t* __restrict__ x1b) {
  __shared__ float Xs[16 * 256];
  __shared__ float redS[256], redQ[256];
  __shared__ float mu_s[16], rs_s[16];

  int tid = threadIdx.x, lane = tid & 31, wv = tid >> 5;
  int r15 = lane & 15, half = lane >> 4;
  int m0 = blockIdx.x * 16;
  int n0 = wv * 32;

  {
    const bf16_t* arow  = attnB + (size_t)(m0 + r15) * Dn + half * 16;
    const bf16_t* brow0 = WoT + (size_t)(n0 + r15) * Dn + half * 16;
    const bf16_t* brow1 = brow0 + (size_t)16 * Dn;
    v8f acc0 = {}, acc1 = {};
    for (int k = 0; k < Dn; k += 32) {
      v16bf a = load_a(arow, k);
      acc0 = wmma_acc(a, load_b(brow0, k), acc0);
      acc1 = wmma_acc(a, load_b(brow1, k), acc1);
    }
#pragma unroll
    for (int r = 0; r < 8; ++r) {
      Xs[(half * 8 + r) * 256 + n0 + r15]      = acc0[r];
      Xs[(half * 8 + r) * 256 + n0 + r15 + 16] = acc1[r];
    }
  }
  __syncthreads();

  int row = tid >> 4, j = tid & 15;
  float vv[16]; float s = 0.f, sq = 0.f;
  for (int i = 0; i < 16; ++i) {
    int c = i * 16 + j;
    float v = Xs[row * 256 + c] + bo[c] + x[((size_t)(m0 + row)) * 256 + c];
    vv[i] = v; s += v; sq += v * v;
  }
  redS[tid] = s; redQ[tid] = sq;
  __syncthreads();
  if (j == 0) {
    float a = 0.f, q = 0.f;
    for (int t = 0; t < 16; ++t) { a += redS[row * 16 + t]; q += redQ[row * 16 + t]; }
    float mu = a / 256.f;
    mu_s[row] = mu;
    rs_s[row] = rsqrtf(q / 256.f - mu * mu + 1e-5f);
  }
  __syncthreads();
  float mu = mu_s[row], rs = rs_s[row];
  for (int i = 0; i < 16; ++i) {
    int c = i * 16 + j;
    float v = (vv[i] - mu) * rs * g1[c] + b1[c];
    size_t idx = ((size_t)(m0 + row)) * 256 + c;
    x1f[idx] = v;
    x1b[idx] = f2bf(v);
  }
}

// ---------------------------------------------------------------------------
// FFN first GEMM + exact GELU.  Wave: 16x32 strip (2 N-tiles, shared A).
// grid: 512 mtiles * 32 pairs / 8 waves = 2048 blocks
// ---------------------------------------------------------------------------
__global__ __launch_bounds__(256) void k_ffn1(
    const bf16_t* __restrict__ x1b, const bf16_t* __restrict__ W1T,
    const float* __restrict__ b1, bf16_t* __restrict__ Hb) {
  int wv = threadIdx.x >> 5, lane = threadIdx.x & 31;
  int r15 = lane & 15, half = lane >> 4;
  int tile = blockIdx.x * 8 + wv;
  int m0 = (tile / 32) * 16;
  int n0 = (tile % 32) * 32;

  const bf16_t* arow  = x1b + (size_t)(m0 + r15) * Dn + half * 16;
  const bf16_t* brow0 = W1T + (size_t)(n0 + r15) * Dn + half * 16;
  const bf16_t* brow1 = brow0 + (size_t)16 * Dn;
  v8f acc0 = {}, acc1 = {};
  for (int k = 0; k < Dn; k += 32) {
    v16bf a = load_a(arow, k);
    acc0 = wmma_acc(a, load_b(brow0, k), acc0);
    acc1 = wmma_acc(a, load_b(brow1, k), acc1);
  }
  int d0 = n0 + r15, d1 = d0 + 16;
  float bb0 = b1[d0], bb1 = b1[d1];
#pragma unroll
  for (int r = 0; r < 8; ++r) {
    size_t rowb = (size_t)(m0 + half * 8 + r) * FFn;
    float v0 = acc0[r] + bb0;
    float v1 = acc1[r] + bb1;
    Hb[rowb + d0] = f2bf(0.5f * v0 * (1.f + erff(v0 * 0.70710678118654752f)));
    Hb[rowb + d1] = f2bf(0.5f * v1 * (1.f + erff(v1 * 0.70710678118654752f)));
  }
}

// ---------------------------------------------------------------------------
// FFN second GEMM + residual + LN2 -> final f32 output
// ---------------------------------------------------------------------------
__global__ __launch_bounds__(256) void k_ffn2_ln2(
    const bf16_t* __restrict__ Hb, const bf16_t* __restrict__ W2T,
    const float* __restrict__ b2, const float* __restrict__ x1f,
    const float* __restrict__ g2, const float* __restrict__ bb2,
    float* __restrict__ out) {
  __shared__ float Xs[16 * 256];
  __shared__ float redS[256], redQ[256];
  __shared__ float mu_s[16], rs_s[16];

  int tid = threadIdx.x, lane = tid & 31, wv = tid >> 5;
  int r15 = lane & 15, half = lane >> 4;
  int m0 = blockIdx.x * 16;
  int n0 = wv * 32;

  {
    const bf16_t* arow  = Hb + (size_t)(m0 + r15) * FFn + half * 16;
    const bf16_t* brow0 = W2T + (size_t)(n0 + r15) * FFn + half * 16;
    const bf16_t* brow1 = brow0 + (size_t)16 * FFn;
    v8f acc0 = {}, acc1 = {};
    for (int k = 0; k < FFn; k += 32) {
      v16bf a = load_a(arow, k);
      acc0 = wmma_acc(a, load_b(brow0, k), acc0);
      acc1 = wmma_acc(a, load_b(brow1, k), acc1);
    }
#pragma unroll
    for (int r = 0; r < 8; ++r) {
      Xs[(half * 8 + r) * 256 + n0 + r15]      = acc0[r];
      Xs[(half * 8 + r) * 256 + n0 + r15 + 16] = acc1[r];
    }
  }
  __syncthreads();

  int row = tid >> 4, j = tid & 15;
  float vv[16]; float s = 0.f, sq = 0.f;
  for (int i = 0; i < 16; ++i) {
    int c = i * 16 + j;
    float v = Xs[row * 256 + c] + b2[c] + x1f[((size_t)(m0 + row)) * 256 + c];
    vv[i] = v; s += v; sq += v * v;
  }
  redS[tid] = s; redQ[tid] = sq;
  __syncthreads();
  if (j == 0) {
    float a = 0.f, q = 0.f;
    for (int t = 0; t < 16; ++t) { a += redS[row * 16 + t]; q += redQ[row * 16 + t]; }
    float mu = a / 256.f;
    mu_s[row] = mu;
    rs_s[row] = rsqrtf(q / 256.f - mu * mu + 1e-5f);
  }
  __syncthreads();
  float mu = mu_s[row], rs = rs_s[row];
  for (int i = 0; i < 16; ++i) {
    int c = i * 16 + j;
    out[((size_t)(m0 + row)) * 256 + c] = (vv[i] - mu) * rs * g2[c] + bb2[c];
  }
}

// ---------------------------------------------------------------------------
extern "C" void kernel_launch(void* const* d_in, const int* in_sizes, int n_in,
                              void* d_out, int out_size, void* d_ws, size_t ws_size,
                              hipStream_t stream) {
  const float*         x    = (const float*)d_in[0];
  const int*           srel = (const int*)d_in[1];
  const unsigned char* kpm  = (const unsigned char*)d_in[2];
  const float *Wq = (const float*)d_in[3],  *bq = (const float*)d_in[4];
  const float *Wk = (const float*)d_in[5],  *bk = (const float*)d_in[6];
  const float *Wv = (const float*)d_in[7],  *bv = (const float*)d_in[8];
  const float *Wo = (const float*)d_in[9],  *bo = (const float*)d_in[10];
  const float *sbw = (const float*)d_in[11];
  const float *g1 = (const float*)d_in[12], *bb1 = (const float*)d_in[13];
  const float *W1 = (const float*)d_in[14], *fb1 = (const float*)d_in[15];
  const float *W2 = (const float*)d_in[16], *fb2 = (const float*)d_in[17];
  const float *g2 = (const float*)d_in[18], *bb2 = (const float*)d_in[19];
  float* out = (float*)d_out;

  const size_t TOK = (size_t)Bn * En;          // 8192 tokens
  char* base = (char*)d_ws;
  size_t off = 0;
  auto alloc = [&](size_t bytes) -> char* {
    char* p = base + off;
    off = (off + bytes + 255) & ~(size_t)255;
    return p;
  };
  bf16_t* xb    = (bf16_t*)alloc(TOK * Dn * 2);        // 4 MB
  bf16_t* WqT   = (bf16_t*)alloc((size_t)Dn * Dn * 2);
  bf16_t* WkT   = (bf16_t*)alloc((size_t)Dn * Dn * 2);
  bf16_t* WvT   = (bf16_t*)alloc((size_t)Dn * Dn * 2);
  bf16_t* WoT   = (bf16_t*)alloc((size_t)Dn * Dn * 2);
  bf16_t* W1T   = (bf16_t*)alloc((size_t)Dn * FFn * 2);
  bf16_t* W2T   = (bf16_t*)alloc((size_t)FFn * Dn * 2);
  bf16_t* Qb    = (bf16_t*)alloc(TOK * Dn * 2);
  bf16_t* Kb    = (bf16_t*)alloc(TOK * Dn * 2);
  bf16_t* Vt    = (bf16_t*)alloc(TOK * Dn * 2);        // [B,H,HD,E]
  bf16_t* attnB = (bf16_t*)alloc(TOK * Dn * 2);
  float*  x1f   = (float*) alloc(TOK * Dn * 4);        // 8 MB
  bf16_t* x1b   = (bf16_t*)alloc(TOK * Dn * 2);
  bf16_t* Hb    = (bf16_t*)alloc(TOK * FFn * 2);       // 16 MB
  (void)ws_size; (void)n_in; (void)in_sizes; (void)out_size;

  // 1) conversions / weight transposes
  {
    int n = (int)(TOK * Dn);
    k_cvt<<<(n + 255) / 256, 256, 0, stream>>>(x, xb, n);
    k_tcvt<<<(Dn * Dn + 255) / 256, 256, 0, stream>>>(Wq, WqT, Dn, Dn);
    k_tcvt<<<(Dn * Dn + 255) / 256, 256, 0, stream>>>(Wk, WkT, Dn, Dn);
    k_tcvt<<<(Dn * Dn + 255) / 256, 256, 0, stream>>>(Wv, WvT, Dn, Dn);
    k_tcvt<<<(Dn * Dn + 255) / 256, 256, 0, stream>>>(Wo, WoT, Dn, Dn);
    k_tcvt<<<(Dn * FFn + 255) / 256, 256, 0, stream>>>(W1, W1T, Dn, FFn);
    k_tcvt<<<(FFn * Dn + 255) / 256, 256, 0, stream>>>(W2, W2T, FFn, Dn);
  }
  // 2) QKV (16x32 strip per wave)
  k_qkv<<<1536, 256, 0, stream>>>(xb, WqT, WkT, WvT, bq, bk, bv, Qb, Kb, Vt);
  // 3) fused attention (99.3 KB dynamic LDS)
  {
    int dyn = 65536 + 32768 + (256 + 16 + 16 + 512 + 48) * 4;
    hipFuncSetAttribute(reinterpret_cast<const void*>(k_attn),
                        hipFuncAttributeMaxDynamicSharedMemorySize, dyn);
    k_attn<<<Bn * Hn * (En / 16), 256, dyn, stream>>>(Qb, Kb, Vt, srel, kpm, sbw, attnB);
  }
  // 4) Wo + residual + LN1
  k_proj_ln1<<<(int)(TOK / 16), 256, 0, stream>>>(attnB, WoT, bo, x, g1, bb1, x1f, x1b);
  // 5) FFN1 + GELU (16x32 strip per wave)
  k_ffn1<<<2048, 256, 0, stream>>>(x1b, W1T, fb1, Hb);
  // 6) FFN2 + residual + LN2
  k_ffn2_ln2<<<(int)(TOK / 16), 256, 0, stream>>>(Hb, W2T, fb2, x1f, g2, bb2, out);
}
